// ImprovedVectorQuantizer_12412455485555
// MI455X (gfx1250) — matrix operand order repaired and checked by reading, not compiled, not run on hardware
//
#include <hip/hip_runtime.h>
#include <hip/hip_bf16.h>

typedef __attribute__((ext_vector_type(16))) __bf16 v16bf;
typedef __attribute__((ext_vector_type(8)))  float  v8f;
typedef unsigned short ushort_t;

union Frag16 { uint4 u[2]; v16bf v; };

#ifndef __has_builtin
#define __has_builtin(x) 0
#endif

#if __has_builtin(__builtin_amdgcn_global_load_async_to_lds_b128)
#define HAVE_ASYNC_LDS 1
#define AS3 __attribute__((address_space(3)))
typedef int v4i_vs __attribute__((vector_size(4 * sizeof(int))));
#else
#define HAVE_ASYNC_LDS 0
#endif

#define NUM_CODES   4096
#define DIM         512
#define N_ROWS      16384
#define N_ELEM      8388608      // 16384*512
#define OUT_IDX_OFF 8388608
#define OUT_LOSS_OFF 8404992

// d_ws layout (bytes)
#define WS_EH   0u
#define WS_EL   (WS_EH + NUM_CODES * DIM * 2u)   // 4 MiB
#define WS_E2   (WS_EL + NUM_CODES * DIM * 2u)   // 8 MiB
#define WS_IDX  (WS_E2 + NUM_CODES * 4u)
#define WS_PART (WS_IDX + N_ROWS * 4u)

// LDS layout for vq_main (dynamic)
// A block: 64 rows x 512 bf16, hi + lo
#define LDS_AH   0
#define LDS_AL   65536
// B double buffer: each buffer = [128 codes][64 bf16] hi (16KB) + lo (16KB)
#define LDS_B0   131072
#define LDS_B1   163840
#define LDS_BLO  16384           // lo-array offset inside a B buffer
#define LDS_REDV 196608
#define LDS_REDI 198656
#define LDS_TOTAL 200704

__device__ __forceinline__ ushort_t f2bf(float f) {
    unsigned u = __float_as_uint(f);
    unsigned r = u + 0x7FFFu + ((u >> 16) & 1u);   // round-to-nearest-even
    return (ushort_t)(r >> 16);
}
__device__ __forceinline__ float bf2f(ushort_t h) {
    return __uint_as_float(((unsigned)h) << 16);
}

__device__ __forceinline__ void waitAsync0() {
#if HAVE_ASYNC_LDS
#if __has_builtin(__builtin_amdgcn_s_wait_asynccnt)
    __builtin_amdgcn_s_wait_asynccnt(0);
#else
    asm volatile("s_wait_asynccnt 0x0" ::: "memory");
#endif
#endif
}

// ---------------------------------------------------------------------------
// Kernel 1: codebook -> bf16 hi/lo split + ||e||^2 per code
// ---------------------------------------------------------------------------
__global__ void __launch_bounds__(256)
vq_prep(const float* __restrict__ cb, ushort_t* __restrict__ eh,
        ushort_t* __restrict__ el, float* __restrict__ e2) {
    __shared__ float red[256];
    const int row = blockIdx.x;
    const int t = threadIdx.x;
    const float* src = cb + (size_t)row * DIM;
    float s = 0.f;
#pragma unroll
    for (int i = 0; i < 2; ++i) {
        int d = t + i * 256;
        float v = src[d];
        ushort_t h = f2bf(v);
        ushort_t l = f2bf(v - bf2f(h));
        eh[(size_t)row * DIM + d] = h;
        el[(size_t)row * DIM + d] = l;
        s += v * v;
    }
    red[t] = s;
    __syncthreads();
    for (int o = 128; o > 0; o >>= 1) {
        if (t < o) red[t] += red[t + o];
        __syncthreads();
    }
    if (t == 0) e2[row] = red[0];
}

// ---------------------------------------------------------------------------
// Kernel 2: fused bf16x3 WMMA GEMM (x . e^T) + argmin over codes
//   block: 256 threads (8 waves), owns 64 rows of x
//   wave w: owns codes (n0 + 16*w .. +16) within each 128-code tile
//   codebook streamed as 128x64 k-chunks, async double-buffered in LDS
// ---------------------------------------------------------------------------
__global__ void __launch_bounds__(256, 1)
vq_main(const float* __restrict__ x, const ushort_t* __restrict__ eh,
        const ushort_t* __restrict__ el, const float* __restrict__ e2,
        int* __restrict__ idx_out, float* __restrict__ idxf_out) {
    extern __shared__ char smem[];
    char* Ah = smem + LDS_AH;
    char* Al = smem + LDS_AL;
    float* redv = (float*)(smem + LDS_REDV);
    int*   redi = (int*)  (smem + LDS_REDI);

    const int t    = threadIdx.x;
    const int lane = t & 31;
    const int wave = t >> 5;
    const int col  = lane & 15;
    const int half = lane >> 4;
    const int mBase = blockIdx.x * 64;

    // B-chunk fill indexing: thread -> (code 0..127, 32-half group 0..1)
    const int cfc = t >> 1;
    const int cfh = t & 1;

    // fill one 128x64-half chunk (hi+lo) into buffer at byte base bbase
    auto fill = [&](int bbase, int n0, int dq) {
        size_t g = (size_t)(n0 + cfc) * DIM + dq + cfh * 32;   // in halves
        const int db = bbase + cfc * 128 + cfh * 64;
#if HAVE_ASYNC_LDS
#pragma unroll
        for (int i = 0; i < 4; ++i) {
            __builtin_amdgcn_global_load_async_to_lds_b128(
                (v4i_vs*)(eh + g + i * 8),
                (AS3 v4i_vs*)(smem + db + i * 16), 0, 0);
            __builtin_amdgcn_global_load_async_to_lds_b128(
                (v4i_vs*)(el + g + i * 8),
                (AS3 v4i_vs*)(smem + db + LDS_BLO + i * 16), 0, 0);
        }
#else
#pragma unroll
        for (int i = 0; i < 4; ++i) {
            uint4 a = *(const uint4*)(eh + g + i * 8);
            uint4 b = *(const uint4*)(el + g + i * 8);
            *(uint4*)(smem + db + i * 16) = a;
            *(uint4*)(smem + db + LDS_BLO + i * 16) = b;
        }
        __builtin_prefetch(eh + g + 64, 0, 1);
        __builtin_prefetch(el + g + 64, 0, 1);
#endif
    };

    // issue fill of chunk 0 first so it overlaps the A-block staging
    fill(LDS_B0, 0, 0);

    // ---- stage x block: fp32 -> bf16 hi/lo into LDS (row-major [64][512]) ----
    const float4* xv4 = (const float4*)(x + (size_t)mBase * DIM);
    for (int j = t; j < (64 * DIM) / 4; j += 256) {
        float4 v = xv4[j];
        ushort_t h0 = f2bf(v.x), h1 = f2bf(v.y), h2 = f2bf(v.z), h3 = f2bf(v.w);
        ushort_t l0 = f2bf(v.x - bf2f(h0)), l1 = f2bf(v.y - bf2f(h1));
        ushort_t l2 = f2bf(v.z - bf2f(h2)), l3 = f2bf(v.w - bf2f(h3));
        uint2 hp, lp;
        hp.x = (unsigned)h0 | ((unsigned)h1 << 16);
        hp.y = (unsigned)h2 | ((unsigned)h3 << 16);
        lp.x = (unsigned)l0 | ((unsigned)l1 << 16);
        lp.y = (unsigned)l2 | ((unsigned)l3 << 16);
        ((uint2*)Ah)[j] = hp;
        ((uint2*)Al)[j] = lp;
    }

    // A fragment byte offsets (ISA 16-bit A layout: half 0 -> K{0..7,16..23})
    const int aSel = half * 16;
    int aOff[4];
#pragma unroll
    for (int m = 0; m < 4; ++m) aOff[m] = (m * 16 + col) * (DIM * 2) + aSel;
    // B fragment (same K order as A: half 0 -> K{0..7,16..23})
    const int codeLocal = wave * 16 + col;

    // running argmin state
    float minv[4][8];
    int   mini[4][8];
#pragma unroll
    for (int m = 0; m < 4; ++m)
#pragma unroll
        for (int r = 0; r < 8; ++r) { minv[m][r] = 3.4e38f; mini[m][r] = 0; }

    for (int n0 = 0; n0 < NUM_CODES; n0 += 128) {
        v8f acc[4];
#pragma unroll
        for (int m = 0; m < 4; ++m)
            acc[m] = (v8f){0.f,0.f,0.f,0.f,0.f,0.f,0.f,0.f};

        for (int dq = 0; dq < DIM; dq += 64) {
            // double-buffer parity depends only on dq (8 chunks per tile, even)
            const int curB  = (dq & 64) ? LDS_B1 : LDS_B0;
            const int nextB = (dq & 64) ? LDS_B0 : LDS_B1;

            waitAsync0();     // this chunk's async fills complete (issued earlier)
            __syncthreads();  // publish chunk; everyone done reading buffer nextB

            // issue next chunk's fill (wraps to next code tile at dq end)
            int nn0 = n0, ndq = dq + 64;
            if (ndq == DIM) { ndq = 0; nn0 += 128; }
            if (nn0 < NUM_CODES) fill(nextB, nn0, ndq);   // overlaps compute

#pragma unroll
            for (int s = 0; s < 2; ++s) {
                const int bb = curB + codeLocal * 128 + s * 64 + half * 16;
                Frag16 bh, bl;
                bh.u[0] = *(const uint4*)(smem + bb);
                bh.u[1] = *(const uint4*)(smem + bb + 32);
                bl.u[0] = *(const uint4*)(smem + bb + LDS_BLO);
                bl.u[1] = *(const uint4*)(smem + bb + LDS_BLO + 32);
#pragma unroll
                for (int m = 0; m < 4; ++m) {
                    Frag16 ah, al;
                    const int ao = aOff[m] + (dq + s * 32) * 2;
                    ah.u[0] = *(const uint4*)(Ah + ao);
                    ah.u[1] = *(const uint4*)(Ah + ao + 32);
                    al.u[0] = *(const uint4*)(Al + ao);
                    al.u[1] = *(const uint4*)(Al + ao + 32);
                    // bf16x3: xh*eh + xh*el + xl*eh  (f32 accumulate)
                    acc[m] = __builtin_amdgcn_wmma_f32_16x16x32_bf16(
                        false, ah.v, false, bh.v, (short)0, acc[m], false, false);
                    acc[m] = __builtin_amdgcn_wmma_f32_16x16x32_bf16(
                        false, ah.v, false, bl.v, (short)0, acc[m], false, false);
                    acc[m] = __builtin_amdgcn_wmma_f32_16x16x32_bf16(
                        false, al.v, false, bh.v, (short)0, acc[m], false, false);
                }
            }
        }

        // epilogue: dist = ||e||^2 - 2*dot ; update running argmin
        const int code = n0 + codeLocal;
        const float e2v = e2[code];
#pragma unroll
        for (int m = 0; m < 4; ++m)
#pragma unroll
            for (int r = 0; r < 8; ++r) {
                float dd = fmaf(-2.0f, acc[m][r], e2v);
                if (dd < minv[m][r]) { minv[m][r] = dd; mini[m][r] = code; }
            }
    }

    // cross-lane reduce within 16-lane halves (C layout: lane%16 == code column)
#pragma unroll
    for (int m = 0; m < 4; ++m)
#pragma unroll
        for (int r = 0; r < 8; ++r) {
            float v = minv[m][r];
            int   i = mini[m][r];
#pragma unroll
            for (int o = 8; o > 0; o >>= 1) {
                float ov = __shfl_xor(v, o, 16);
                int   oi = __shfl_xor(i, o, 16);
                if (ov < v || (ov == v && oi < i)) { v = ov; i = oi; }
            }
            if (col == 0) {
                const int rowLocal = m * 16 + r + 8 * half;
                redv[wave * 64 + rowLocal] = v;
                redi[wave * 64 + rowLocal] = i;
            }
        }
    __syncthreads();

    // cross-wave reduce (8 waves held disjoint code subsets of every tile)
    if (t < 64) {
        float v = redv[t];
        int   i = redi[t];
#pragma unroll
        for (int w = 1; w < 8; ++w) {
            float ov = redv[w * 64 + t];
            int   oi = redi[w * 64 + t];
            if (ov < v || (ov == v && oi < i)) { v = ov; i = oi; }
        }
        const int row = mBase + t;
        idx_out[row]  = i;
        idxf_out[row] = (float)i;
    }
}

// ---------------------------------------------------------------------------
// Kernel 3: gather quantized rows, straight-through output, per-block SSE
// ---------------------------------------------------------------------------
__global__ void __launch_bounds__(256)
vq_gather(const float* __restrict__ x, const float* __restrict__ cb,
          const int* __restrict__ idx, float* __restrict__ out,
          float* __restrict__ part) {
    __shared__ float red[256];
    const int t = threadIdx.x;
    const size_t base = (size_t)blockIdx.x * 8192 + t;
    float s = 0.f;
#pragma unroll
    for (int i = 0; i < 32; ++i) {
        size_t e = base + (size_t)i * 256;
        int row = (int)(e >> 9);
        int d   = (int)(e & 511);
        int k   = idx[row];
        float q  = cb[(size_t)k * DIM + d];
        float xv = x[e];
        float diff = q - xv;
        out[e] = xv + diff;        // straight-through, reference rounding order
        s += diff * diff;
    }
    red[t] = s;
    __syncthreads();
    for (int o = 128; o > 0; o >>= 1) {
        if (t < o) red[t] += red[t + o];
        __syncthreads();
    }
    if (t == 0) part[blockIdx.x] = red[0];
}

// ---------------------------------------------------------------------------
// Kernel 4: final loss reduction (deterministic, no atomics)
// ---------------------------------------------------------------------------
__global__ void __launch_bounds__(256)
vq_loss(const float* __restrict__ part, float* __restrict__ loss_out) {
    __shared__ float red[256];
    const int t = threadIdx.x;
    float s = part[t] + part[t + 256] + part[t + 512] + part[t + 768];
    red[t] = s;
    __syncthreads();
    for (int o = 128; o > 0; o >>= 1) {
        if (t < o) red[t] += red[t + o];
        __syncthreads();
    }
    if (t == 0) {
        float m = red[0] * (1.0f / (float)N_ELEM);
        loss_out[0] = 0.25f * m;   // commitment loss
        loss_out[1] = m;           // codebook loss (same mean numerically)
    }
}

// ---------------------------------------------------------------------------
extern "C" void kernel_launch(void* const* d_in, const int* in_sizes, int n_in,
                              void* d_out, int out_size, void* d_ws, size_t ws_size,
                              hipStream_t stream) {
    (void)in_sizes; (void)n_in; (void)out_size; (void)ws_size;
    const float* x  = (const float*)d_in[0];
    const float* cb = (const float*)d_in[1];
    float* out = (float*)d_out;

    char* ws = (char*)d_ws;
    ushort_t* eh  = (ushort_t*)(ws + WS_EH);
    ushort_t* el  = (ushort_t*)(ws + WS_EL);
    float*    e2  = (float*)(ws + WS_E2);
    int*      idx = (int*)(ws + WS_IDX);
    float*    part= (float*)(ws + WS_PART);

    // allow >64KB dynamic LDS (idempotent; not a stream op, capture-safe)
    (void)hipFuncSetAttribute(reinterpret_cast<const void*>(vq_main),
                              hipFuncAttributeMaxDynamicSharedMemorySize,
                              LDS_TOTAL);

    vq_prep  <<<NUM_CODES, 256, 0, stream>>>(cb, eh, el, e2);
    vq_main  <<<N_ROWS / 64, 256, LDS_TOTAL, stream>>>(x, eh, el, e2, idx,
                                                       out + OUT_IDX_OFF);
    vq_gather<<<1024, 256, 0, stream>>>(x, cb, idx, out, part);
    vq_loss  <<<1, 256, 0, stream>>>(part, out + OUT_LOSS_OFF);
}
